// QFWEncoderLayer_28381143892479
// MI455X (gfx1250) — compile-verified, matrix operand order checked by reading, not run on hardware
//
#include <hip/hip_runtime.h>

#define DEV __device__ __forceinline__

constexpr int kB  = 4;
constexpr int kS  = 2048;
constexpr int kD  = 1024;
constexpr int kH  = 16;
constexpr int kF  = 4096;
constexpr int kHD = 64;
constexpr int kM  = kB * kS;  // 8192 rows

typedef __attribute__((ext_vector_type(16))) _Float16 v16h;
typedef __attribute__((ext_vector_type(8)))  _Float16 v8h;
typedef __attribute__((ext_vector_type(8)))  float    v8f;

// ---------------------------------------------------------------------------
// helpers
// ---------------------------------------------------------------------------
DEV float qdq_f(float v, float s) {
  float q = rintf(v / s) + 64.0f;           // quint8 zero-point 64
  q = fminf(fmaxf(q, 0.0f), 255.0f);
  return (q - 64.0f) * s;
}

// Build a 16-bit A/B WMMA fragment from a K-contiguous row.
// CDNA5 layout: lanes 0-15 hold K chunks {0..7, 16..23}; lanes 16-31 hold
// {8..15, 24..31}.  g = lane>>4 selects the chunk pair; loads are 16B each.
DEV v16h load_frag16(const _Float16* p, int g) {
  const v8h* q = (const v8h*)p;
  v8h lo = q[g];        // chunk 0 or 1
  v8h hi = q[g + 2];    // chunk 2 or 3
  v16h r;
#pragma unroll
  for (int i = 0; i < 8; ++i) { r[i] = lo[i]; r[i + 8] = hi[i]; }
  return r;
}

DEV v8f wmma_f16(v16h a, v16h b, v8f c) {
  return __builtin_amdgcn_wmma_f32_16x16x32_f16(false, a, false, b,
                                                (short)0, c, false, false);
}

// ---------------------------------------------------------------------------
// elementwise / reduction kernels
// ---------------------------------------------------------------------------
__global__ void init_scalars_kernel(unsigned int* scal) {
  if (threadIdx.x < 8) scal[threadIdx.x] = 0u;
}

__global__ __launch_bounds__(256)
void absmax_kernel(const float* __restrict__ x, size_t n, unsigned int* out) {
  float m = 0.0f;
  size_t stride = (size_t)gridDim.x * blockDim.x;
  for (size_t i = (size_t)blockIdx.x * blockDim.x + threadIdx.x; i < n; i += stride)
    m = fmaxf(m, fabsf(x[i]));
#pragma unroll
  for (int k = 1; k < 32; k <<= 1) m = fmaxf(m, __shfl_xor(m, k));
  __shared__ float red[8];
  int wv = threadIdx.x >> 5, lane = threadIdx.x & 31;
  if (lane == 0) red[wv] = m;
  __syncthreads();
  if (threadIdx.x == 0) {
    float mm = 0.0f;
#pragma unroll
    for (int i = 0; i < 8; ++i) mm = fmaxf(mm, red[i]);
    atomicMax(out, __float_as_uint(mm));   // positive floats order as uints
  }
}

__global__ __launch_bounds__(256)
void qdq_dyn_kernel(const float* __restrict__ x, const unsigned int* __restrict__ amax,
                    size_t n, float* __restrict__ outF, _Float16* __restrict__ outH) {
  float s = __uint_as_float(*amax) / 127.0f + 1e-12f;
  size_t stride = (size_t)gridDim.x * blockDim.x;
  for (size_t i = (size_t)blockIdx.x * blockDim.x + threadIdx.x; i < n; i += stride) {
    float v = qdq_f(x[i], s);
    if (outF) outF[i] = v;
    if (outH) outH[i] = (_Float16)v;
  }
}

__global__ __launch_bounds__(256)
void add_qdq_kernel(const float* __restrict__ a, const float* __restrict__ b,
                    const float* __restrict__ opsc, int sidx, size_t n,
                    float* __restrict__ outF) {
  float s = opsc[sidx] * 0.05f;
  size_t stride = (size_t)gridDim.x * blockDim.x;
  for (size_t i = (size_t)blockIdx.x * blockDim.x + threadIdx.x; i < n; i += stride)
    outF[i] = qdq_f(a[i] + b[i], s);
}

__global__ __launch_bounds__(256)
void f32_to_f16_kernel(const float* __restrict__ in, _Float16* __restrict__ out, size_t n) {
  size_t stride = (size_t)gridDim.x * blockDim.x;
  for (size_t i = (size_t)blockIdx.x * blockDim.x + threadIdx.x; i < n; i += stride)
    out[i] = (_Float16)in[i];
}

// v [B,S,D] (head-major within D) -> vT [B,H,HD,S]  (K-contiguous for WMMA B)
__global__ __launch_bounds__(256)
void transpose_v_kernel(const _Float16* __restrict__ v, _Float16* __restrict__ vT) {
  size_t n = (size_t)kM * kD;
  size_t stride = (size_t)gridDim.x * blockDim.x;
  for (size_t o = (size_t)blockIdx.x * blockDim.x + threadIdx.x; o < n; o += stride) {
    int s_ = (int)(o % kS);
    int d  = (int)((o / kS) % kHD);
    int h  = (int)((o / ((size_t)kS * kHD)) % kH);
    int b  = (int)(o / ((size_t)kS * kHD * kH));
    vT[o] = v[((size_t)(b * kS + s_)) * kD + h * kHD + d];
  }
}

// ---------------------------------------------------------------------------
// WMMA GEMM:  out[M,N] = A[M,K] @ Bw[N,K]^T + bias, optional relu + qdq
// one wave computes a 64x64 C tile (16 accumulators); K-loop step 32.
// 16 v_wmma per 8 fragment loads -> 2x the arithmetic intensity of a 32x32
// tile, halving L2 traffic for the FFN GEMMs that dominate the FLOP count.
// ---------------------------------------------------------------------------
__global__ __launch_bounds__(256)
void gemm_wmma_kernel(const _Float16* __restrict__ A, const _Float16* __restrict__ Bw,
                      const float* __restrict__ bias, const float* __restrict__ opsc,
                      int sidx, int relu, _Float16* __restrict__ outH,
                      float* __restrict__ outF, int M, int N, int K) {
  int wave   = (int)((blockIdx.x * blockDim.x + threadIdx.x) >> 5);
  int lane   = threadIdx.x & 31;
  int tilesN = N >> 6;
  int tilesM = M >> 6;
  if (wave >= tilesM * tilesN) return;           // wave-uniform: EXEC stays full
  int tm = (wave / tilesN) << 6;
  int tn = (wave % tilesN) << 6;
  int g  = lane >> 4;
  int ln = lane & 15;

  v8f acc[4][4] = {};
  const _Float16* Arow[4];
  const _Float16* Brow[4];
#pragma unroll
  for (int t = 0; t < 4; ++t) {
    Arow[t] = A  + (size_t)(tm + t * 16 + ln) * K;
    Brow[t] = Bw + (size_t)(tn + t * 16 + ln) * K;
  }

  for (int k = 0; k < K; k += 32) {
    int pr = (k >> 5) & 3;                        // rotate prefetched row
    __builtin_prefetch(Arow[pr] + k + 256, 0, 3); // near-cache prefetch
    __builtin_prefetch(Brow[pr] + k + 256, 0, 3);
    v16h af[4], bf[4];
#pragma unroll
    for (int t = 0; t < 4; ++t) af[t] = load_frag16(Arow[t] + k, g);
#pragma unroll
    for (int t = 0; t < 4; ++t) bf[t] = load_frag16(Brow[t] + k, g);
#pragma unroll
    for (int i = 0; i < 4; ++i)
#pragma unroll
      for (int j = 0; j < 4; ++j)
        acc[i][j] = wmma_f16(af[i], bf[j], acc[i][j]);
  }

  float s = (sidx >= 0) ? opsc[sidx] * 0.05f : 1.0f;
#pragma unroll
  for (int i = 0; i < 4; ++i) {
#pragma unroll
    for (int j = 0; j < 4; ++j) {
      int col = tn + j * 16 + ln;
      float bv = bias ? bias[col] : 0.0f;
#pragma unroll
      for (int r = 0; r < 8; ++r) {
        int row = tm + i * 16 + r + (g << 3);    // VGPR r: rows r / r+8 per half-wave
        float v = acc[i][j][r] + bv;
        if (relu) v = fmaxf(v, 0.0f);
        if (sidx >= 0) v = qdq_f(v, s);
        size_t idx = (size_t)row * N + col;
        if (outF) outF[idx] = v;
        if (outH) outH[idx] = (_Float16)v;
      }
    }
  }
}

// ---------------------------------------------------------------------------
// Flash attention: block = 4 waves, each wave owns 16 q rows of one (b,h).
// scores via WMMA (HD=64 -> 2 k-steps), online softmax, P restaged through
// LDS (C-layout -> A-layout), P@V via WMMA against pre-transposed vT.
// ---------------------------------------------------------------------------
__global__ __launch_bounds__(128)
void flash_attn_kernel(const _Float16* __restrict__ Qh, const _Float16* __restrict__ Kh,
                       const _Float16* __restrict__ vT, const unsigned char* __restrict__ mask,
                       float* __restrict__ ctx) {
  __shared__ _Float16 lds[4][16 * 32];
  int wv   = threadIdx.x >> 5;
  int lane = threadIdx.x & 31;
  int g = lane >> 4, ln = lane & 15;

  int qPerBH = kS / 64;
  int qBlock = blockIdx.x % qPerBH;
  int bh     = blockIdx.x / qPerBH;
  int h = bh % kH, b = bh / kH;
  int q0 = qBlock * 64 + wv * 16;

  size_t qbase = ((size_t)(b * kS + q0 + ln)) * kD + h * kHD;
  v16h a0 = load_frag16(Qh + qbase, g);        // d = 0..31
  v16h a1 = load_frag16(Qh + qbase + 32, g);   // d = 32..63

  v8f oacc[4] = {};
  float mrow[8], lrow[8];
#pragma unroll
  for (int r = 0; r < 8; ++r) { mrow[r] = -1e30f; lrow[r] = 0.0f; }

  for (int kb = 0; kb < kS; kb += 32) {
    v8f z = {};
    size_t k0 = ((size_t)(b * kS + kb + ln)) * kD + h * kHD;
    size_t k1 = ((size_t)(b * kS + kb + 16 + ln)) * kD + h * kHD;
    v16h b00 = load_frag16(Kh + k0, g);
    v16h b01 = load_frag16(Kh + k0 + 32, g);
    v8f p0 = wmma_f16(a0, b00, z);
    p0 = wmma_f16(a1, b01, p0);
    v16h b10 = load_frag16(Kh + k1, g);
    v16h b11 = load_frag16(Kh + k1 + 32, g);
    v8f p1 = wmma_f16(a0, b10, z);
    p1 = wmma_f16(a1, b11, p1);

    unsigned char m0 = mask[(size_t)b * kS + kb + ln];
    unsigned char m1 = mask[(size_t)b * kS + kb + 16 + ln];
#pragma unroll
    for (int r = 0; r < 8; ++r) {
      p0[r] = m0 ? p0[r] * 0.125f : -1e9f;     // 1/sqrt(HD)
      p1[r] = m1 ? p1[r] * 0.125f : -1e9f;
    }

    // online softmax per row (rows live per-VGPR; reduce over 16-lane group)
#pragma unroll
    for (int r = 0; r < 8; ++r) {
      float tm = fmaxf(p0[r], p1[r]);
      tm = fmaxf(tm, __shfl_xor(tm, 1));
      tm = fmaxf(tm, __shfl_xor(tm, 2));
      tm = fmaxf(tm, __shfl_xor(tm, 4));
      tm = fmaxf(tm, __shfl_xor(tm, 8));
      float mn   = fmaxf(mrow[r], tm);
      float corr = expf(mrow[r] - mn);
      p0[r] = expf(p0[r] - mn);
      p1[r] = expf(p1[r] - mn);
      float rs = p0[r] + p1[r];
      rs += __shfl_xor(rs, 1);
      rs += __shfl_xor(rs, 2);
      rs += __shfl_xor(rs, 4);
      rs += __shfl_xor(rs, 8);
      lrow[r] = lrow[r] * corr + rs;
      mrow[r] = mn;
#pragma unroll
      for (int dc = 0; dc < 4; ++dc) oacc[dc][r] *= corr;
    }

    // restage P (C-layout f32) -> LDS -> A-layout f16 fragment
    __syncthreads();
#pragma unroll
    for (int r = 0; r < 8; ++r) {
      int rr = r + (g << 3);
      lds[wv][rr * 32 + ln]      = (_Float16)p0[r];
      lds[wv][rr * 32 + 16 + ln] = (_Float16)p1[r];
    }
    __syncthreads();
    v16h pa = load_frag16(&lds[wv][ln * 32], g);

    // P @ V : vT is [B,H,HD,S], K-contiguous per d column
#pragma unroll
    for (int dc = 0; dc < 4; ++dc) {
      size_t vb = (((size_t)(b * kH + h) * kHD) + dc * 16 + ln) * kS + kb;
      v16h bf = load_frag16(vT + vb, g);
      oacc[dc] = wmma_f16(pa, bf, oacc[dc]);
    }
  }

#pragma unroll
  for (int r = 0; r < 8; ++r) {
    int q = q0 + r + (g << 3);
    float inv = 1.0f / fmaxf(lrow[r], 1e-20f);
    size_t obase = ((size_t)(b * kS + q)) * kD + h * kHD;
#pragma unroll
    for (int dc = 0; dc < 4; ++dc)
      ctx[obase + dc * 16 + ln] = oacc[dc][r] * inv;
  }
}

// ---------------------------------------------------------------------------
// LayerNorm(+qdq): one block per row of D=1024
// ---------------------------------------------------------------------------
__global__ __launch_bounds__(256)
void layernorm_qdq_kernel(const float* __restrict__ in, const float* __restrict__ w,
                          const float* __restrict__ bb, const float* __restrict__ opsc,
                          int sidx, float* __restrict__ outF, _Float16* __restrict__ outH) {
  const float* x = in + (size_t)blockIdx.x * kD;
  float s1 = 0.0f, s2 = 0.0f;
  for (int i = threadIdx.x; i < kD; i += 256) {
    float v = x[i];
    s1 += v; s2 += v * v;
  }
#pragma unroll
  for (int k = 1; k < 32; k <<= 1) { s1 += __shfl_xor(s1, k); s2 += __shfl_xor(s2, k); }
  __shared__ float red[16];
  int wv = threadIdx.x >> 5, lane = threadIdx.x & 31;
  if (lane == 0) { red[wv] = s1; red[8 + wv] = s2; }
  __syncthreads();
  float t1 = 0.0f, t2 = 0.0f;
#pragma unroll
  for (int i = 0; i < 8; ++i) { t1 += red[i]; t2 += red[8 + i]; }
  float mu   = t1 / kD;
  float var  = t2 / kD - mu * mu;
  float rstd = rsqrtf(var + 1e-5f);
  float s = opsc[sidx] * 0.05f;
  for (int i = threadIdx.x; i < kD; i += 256) {
    float v = (x[i] - mu) * rstd * w[i] + bb[i];
    v = qdq_f(v, s);
    size_t idx = (size_t)blockIdx.x * kD + i;
    if (outF) outF[idx] = v;
    if (outH) outH[idx] = (_Float16)v;
  }
}

// ---------------------------------------------------------------------------
// launch
// ---------------------------------------------------------------------------
extern "C" void kernel_launch(void* const* d_in, const int* in_sizes, int n_in,
                              void* d_out, int out_size, void* d_ws, size_t ws_size,
                              hipStream_t stream) {
  (void)in_sizes; (void)n_in; (void)out_size; (void)ws_size;
  const float* x    = (const float*)d_in[0];
  const unsigned char* mask = (const unsigned char*)d_in[1];
  const float* wq   = (const float*)d_in[2];
  const float* bq   = (const float*)d_in[3];
  const float* wk   = (const float*)d_in[4];
  const float* bk   = (const float*)d_in[5];
  const float* wv   = (const float*)d_in[6];
  const float* bv   = (const float*)d_in[7];
  const float* wo   = (const float*)d_in[8];
  const float* bo   = (const float*)d_in[9];
  const float* w1   = (const float*)d_in[10];
  const float* b1   = (const float*)d_in[11];
  const float* w2   = (const float*)d_in[12];
  const float* b2   = (const float*)d_in[13];
  const float* ln1w = (const float*)d_in[14];
  const float* ln1b = (const float*)d_in[15];
  const float* ln2w = (const float*)d_in[16];
  const float* ln2b = (const float*)d_in[17];
  const float* opsc = (const float*)d_in[18];
  float* out = (float*)d_out;

  const size_t nMD = (size_t)kM * kD;   // 8.4M
  const size_t nMF = (size_t)kM * kF;   // 33.5M
  char* ws = (char*)d_ws;
  size_t off = 0;
  auto alloc = [&](size_t bytes) -> char* {
    char* p = ws + off;
    off += (bytes + 255) & ~(size_t)255;
    return p;
  };
  unsigned int* scal = (unsigned int*)alloc(256);   // [0]=absmax(x) [1]=absmax(ctx)
  float*    xq32  = (float*)   alloc(nMD * 4);
  _Float16* xq16  = (_Float16*)alloc(nMD * 2);
  _Float16* q16   = (_Float16*)alloc(nMD * 2);
  _Float16* k16   = (_Float16*)alloc(nMD * 2);
  _Float16* v16   = (_Float16*)alloc(nMD * 2);
  _Float16* vT16  = (_Float16*)alloc(nMD * 2);
  float*    ctx32 = (float*)   alloc(nMD * 4);
  _Float16* ctx16 = (_Float16*)alloc(nMD * 2);
  float*    o32   = (float*)   alloc(nMD * 4);
  float*    h132  = (float*)   alloc(nMD * 4);
  float*    h32   = (float*)   alloc(nMD * 4);
  _Float16* h16   = (_Float16*)alloc(nMD * 2);
  _Float16* f16b  = (_Float16*)alloc(nMF * 2);
  float*    f232  = (float*)   alloc(nMD * 4);
  float*    t232  = (float*)   alloc(nMD * 4);
  _Float16* wq16p = (_Float16*)alloc((size_t)kD * kD * 2);
  _Float16* wk16p = (_Float16*)alloc((size_t)kD * kD * 2);
  _Float16* wv16p = (_Float16*)alloc((size_t)kD * kD * 2);
  _Float16* wo16p = (_Float16*)alloc((size_t)kD * kD * 2);
  _Float16* w116p = (_Float16*)alloc((size_t)kF * kD * 2);
  _Float16* w216p = (_Float16*)alloc((size_t)kD * kF * 2);

  // 1) dynamic input scale + qdq(x)
  init_scalars_kernel<<<1, 32, 0, stream>>>(scal);
  absmax_kernel<<<4096, 256, 0, stream>>>(x, nMD, scal + 0);
  qdq_dyn_kernel<<<8192, 256, 0, stream>>>(x, scal + 0, nMD, xq32, xq16);

  // 2) weights -> f16
  f32_to_f16_kernel<<<2048, 256, 0, stream>>>(wq, wq16p, (size_t)kD * kD);
  f32_to_f16_kernel<<<2048, 256, 0, stream>>>(wk, wk16p, (size_t)kD * kD);
  f32_to_f16_kernel<<<2048, 256, 0, stream>>>(wv, wv16p, (size_t)kD * kD);
  f32_to_f16_kernel<<<2048, 256, 0, stream>>>(wo, wo16p, (size_t)kD * kD);
  f32_to_f16_kernel<<<4096, 256, 0, stream>>>(w1, w116p, (size_t)kF * kD);
  f32_to_f16_kernel<<<4096, 256, 0, stream>>>(w2, w216p, (size_t)kD * kF);

  // 3) QKV projections (note reference wiring: k<-wv/bv, v<-wk/bk)
  int blkDD = ((kM / 64) * (kD / 64)) / 8;   // 256 blocks of 8 waves
  gemm_wmma_kernel<<<blkDD, 256, 0, stream>>>(xq16, wq16p, bq, opsc, 0, 0, q16, nullptr, kM, kD, kD);
  gemm_wmma_kernel<<<blkDD, 256, 0, stream>>>(xq16, wv16p, bv, opsc, 1, 0, k16, nullptr, kM, kD, kD);
  gemm_wmma_kernel<<<blkDD, 256, 0, stream>>>(xq16, wk16p, bk, opsc, 2, 0, v16, nullptr, kM, kD, kD);

  // 4) attention
  transpose_v_kernel<<<8192, 256, 0, stream>>>(v16, vT16);
  flash_attn_kernel<<<kB * kH * (kS / 64), 128, 0, stream>>>(q16, k16, vT16, mask, ctx32);

  // 5) requantize ctx (dynamic) + out projection
  absmax_kernel<<<4096, 256, 0, stream>>>(ctx32, nMD, scal + 1);
  qdq_dyn_kernel<<<8192, 256, 0, stream>>>(ctx32, scal + 1, nMD, nullptr, ctx16);
  gemm_wmma_kernel<<<blkDD, 256, 0, stream>>>(ctx16, wo16p, bo, opsc, 3, 0, nullptr, o32, kM, kD, kD);

  // 6) residual + LN1
  add_qdq_kernel<<<8192, 256, 0, stream>>>(xq32, o32, opsc, 8, nMD, h132);
  layernorm_qdq_kernel<<<kM, 256, 0, stream>>>(h132, ln1w, ln1b, opsc, 6, h32, h16);

  // 7) FFN
  int blkDF = ((kM / 64) * (kF / 64)) / 8;   // 1024 blocks
  gemm_wmma_kernel<<<blkDF, 256, 0, stream>>>(h16, w116p, b1, opsc, 4, 1, f16b, nullptr, kM, kF, kD);
  gemm_wmma_kernel<<<blkDD, 256, 0, stream>>>(f16b, w216p, b2, opsc, 5, 0, nullptr, f232, kM, kD, kF);

  // 8) residual + LN2 -> output
  add_qdq_kernel<<<8192, 256, 0, stream>>>(h32, f232, opsc, 9, nMD, t232);
  layernorm_qdq_kernel<<<kM, 256, 0, stream>>>(t232, ln2w, ln2b, opsc, 7, out, nullptr);
}
